// TemplatePointwiseAttention_2199023255616
// MI455X (gfx1250) — matrix operand order repaired and checked by reading, not compiled, not run on hardware
//
#include <hip/hip_runtime.h>

// ---------------------------------------------------------------------------
// TemplatePointwiseAttention for MI455X (gfx1250, wave32, WMMA 16x16x32 f16)
//
// Shapes: z2d (65536 x 128), t2d (4 x 65536 x 128), Wq/Wk/Wv (256 x 128),
//         Wo (128 x 256), bo (128), out (65536 x 128).
// Channel o of the 256-wide projection maps to (c, h) as o = c*4 + h.
//
// v2: 32-row tiles (2 M-tiles), Wk/Wv B-fragments hoisted across the
//     template loop to cut L2 weight-fragment traffic ~5x.
// ---------------------------------------------------------------------------

typedef __attribute__((ext_vector_type(16))) _Float16 v16h;
typedef __attribute__((ext_vector_type(8)))  _Float16 v8h;
typedef __attribute__((ext_vector_type(8)))  float    v8f;

#define C_IN   128
#define C_QK   256
#define TPL    4
#define ROWS   32          // positions per workgroup tile
#define MT     2           // 16-row M-tiles per workgroup
#define NPOS   65536       // B*I*J
#define ZST    136         // f16 row stride for staged A tiles (bank-spread, 16B-aligned)
#define QST    258         // f32 row stride for Q/K score buffers
#define OST    264         // f16 row stride for Out tile (multiple of 8)

// ---- fragment helpers -----------------------------------------------------
// A fragment (16x32 MxK, 16-bit): lane<16 row m holds K {k..k+15};
// lane>=16 holds K {k+16..k+31}.  Two 16B chunks per lane.
__device__ __forceinline__ v16h load_afrag(const _Float16* base, int rs,
                                           int m, int kbase, int half) {
  const _Float16* r = base + m * rs + kbase + half * 16;
  v8h lo = *(const v8h*)(r);
  v8h hi = *(const v8h*)(r + 8);
  return __builtin_shufflevector(lo, hi, 0,1,2,3,4,5,6,7,8,9,10,11,12,13,14,15);
}

// B fragment (32x16 KxN, 16-bit): lane n (0-15) holds K=k..k+15, lane 16-31
// holds K=k+16..k+31 for column n-16; 32 contiguous bytes per lane when the
// weight matrix is stored K-contiguous ([n][k]).
__device__ __forceinline__ v16h load_bfrag(const _Float16* g) {
  v8h lo = *(const v8h*)(g);
  v8h hi = *(const v8h*)(g + 8);
  return __builtin_shufflevector(lo, hi, 0,1,2,3,4,5,6,7,8,9,10,11,12,13,14,15);
}

__device__ __forceinline__ v8f wmma16(v16h a, v16h b, v8f c) {
  // D = A*B + C,  16x16x32 f16 -> f32 accumulate
  return __builtin_amdgcn_wmma_f32_16x16x32_f16(false, a, false, b,
                                                (short)0, c, false, false);
}

__device__ __forceinline__ v8h cvt8(float4 f0, float4 f1) {
  v8h h;
  h[0]=(_Float16)f0.x; h[1]=(_Float16)f0.y; h[2]=(_Float16)f0.z; h[3]=(_Float16)f0.w;
  h[4]=(_Float16)f1.x; h[5]=(_Float16)f1.y; h[6]=(_Float16)f1.z; h[7]=(_Float16)f1.w;
  return h;
}

// ---- weight f32 -> f16 prep ----------------------------------------------
__global__ void cvt_f32_f16(const float* __restrict__ s, _Float16* __restrict__ d, int n) {
  int i = blockIdx.x * 256 + threadIdx.x;
  if (i < n) d[i] = (_Float16)s[i];
}

// ---- fused attention kernel ----------------------------------------------
__global__ void __launch_bounds__(256)
tpa_main(const float* __restrict__ z2d, const float* __restrict__ t2d,
         const _Float16* __restrict__ Wq_h, const _Float16* __restrict__ Wk_h,
         const _Float16* __restrict__ Wv_h, const _Float16* __restrict__ Wo_h,
         const float* __restrict__ bo, float* __restrict__ out) {
  __shared__ __align__(16) _Float16 Zs[ROWS * ZST];            //  8.7 KB
  __shared__ __align__(16) _Float16 Ts[TPL][ROWS * ZST];       // 34.8 KB
  __shared__ __align__(16) float    Qs[ROWS * QST];            // 33.0 KB
  __shared__ __align__(16) float    Ks[ROWS * QST];            // 33.0 KB
  __shared__ float Sc[ROWS][TPL][4];                           //  2 KB
  __shared__ float Wb[ROWS][TPL][4];                           //  2 KB
  __shared__ __align__(16) _Float16 Os[ROWS * OST];            // 16.9 KB

  const int tid  = threadIdx.x;
  const int lane = tid & 31;
  const int wave = tid >> 5;          // 0..7
  const int half = (lane >> 4) & 1;
  const int lm   = lane & 15;
  const int p0   = blockIdx.x * ROWS;

  // ---- stage Z and T tiles: global f32 -> LDS f16 (2 rows per thread) ----
#pragma unroll
  for (int r2 = 0; r2 < 2; ++r2) {
    int m = (tid >> 4) + r2 * 16, seg = tid & 15;
    const float* zr = z2d + (size_t)(p0 + m) * C_IN + seg * 8;
    *(v8h*)&Zs[m * ZST + seg * 8] = cvt8(*(const float4*)zr, *(const float4*)(zr + 4));
#pragma unroll
    for (int t = 0; t < TPL; ++t) {
      const float* tr = t2d + ((size_t)t * NPOS + p0 + m) * C_IN + seg * 8;
      *(v8h*)&Ts[t][m * ZST + seg * 8] =
          cvt8(*(const float4*)tr, *(const float4*)(tr + 4));
    }
  }
  __syncthreads();

  // ---- GEMM1: Q (32x256) = Z (32x128) x Wq^T; each wave: 2 n-tiles x 2 m ----
  {
    v16h az[MT][4];
#pragma unroll
    for (int mt = 0; mt < MT; ++mt)
#pragma unroll
      for (int k = 0; k < 4; ++k)
        az[mt][k] = load_afrag(Zs + mt * 16 * ZST, ZST, lm, k * 32, half);

#pragma unroll
    for (int i = 0; i < 2; ++i) {
      int nt = wave * 2 + i;
      v8f acc[MT];
      acc[0] = {}; acc[1] = {};
#pragma unroll
      for (int k = 0; k < 4; ++k) {
        v16h b = load_bfrag(Wq_h + (size_t)(nt * 16 + lm) * C_IN + k * 32 + half * 16);
#pragma unroll
        for (int mt = 0; mt < MT; ++mt) acc[mt] = wmma16(az[mt][k], b, acc[mt]);
      }
#pragma unroll
      for (int mt = 0; mt < MT; ++mt)
#pragma unroll
        for (int v = 0; v < 8; ++v)
          Qs[(mt * 16 + v + half * 8) * QST + nt * 16 + lm] = acc[mt][v];
    }
  }
  __syncthreads();

  // ---- per-template: K_t GEMM (hoisted Wk frags) + per-(row,head) dots ----
  {
    v16h bk[2][4];                      // this wave's Wk fragments, reused 4x
#pragma unroll
    for (int i = 0; i < 2; ++i)
#pragma unroll
      for (int k = 0; k < 4; ++k)
        bk[i][k] = load_bfrag(Wk_h + (size_t)((wave * 2 + i) * 16 + lm) * C_IN +
                              k * 32 + half * 16);

    for (int t = 0; t < TPL; ++t) {
      v16h at[MT][4];
#pragma unroll
      for (int mt = 0; mt < MT; ++mt)
#pragma unroll
        for (int k = 0; k < 4; ++k)
          at[mt][k] = load_afrag(Ts[t] + mt * 16 * ZST, ZST, lm, k * 32, half);
#pragma unroll
      for (int i = 0; i < 2; ++i) {
        int nt = wave * 2 + i;
        v8f acc[MT];
        acc[0] = {}; acc[1] = {};
#pragma unroll
        for (int k = 0; k < 4; ++k)
#pragma unroll
          for (int mt = 0; mt < MT; ++mt) acc[mt] = wmma16(at[mt][k], bk[i][k], acc[mt]);
#pragma unroll
        for (int mt = 0; mt < MT; ++mt)
#pragma unroll
          for (int v = 0; v < 8; ++v)
            Ks[(mt * 16 + v + half * 8) * QST + nt * 16 + lm] = acc[mt][v];
      }
      __syncthreads();

      // scores[row][t][h] = (1/sqrt(128)) * sum_c Q[row][4c+h]*K[row][4c+h]
      {
        int row = tid >> 3, h = (tid >> 1) & 3, g = tid & 1;  // c = g + 2j
        float part = 0.f;
#pragma unroll
        for (int j = 0; j < 32; ++j) {
          int o = 4 * (g + 2 * j) + h;
          part += Qs[row * QST + o] * Ks[row * QST + o];
        }
        part += __shfl_xor(part, 1, 32);
        if (g == 0) Sc[row][t][h] = part * 0.08838834764831845f;  // 1/sqrt(128)
      }
      __syncthreads();
    }
  }

  // ---- softmax over the 4 templates ----
  if (tid < 128) {
    int row = tid >> 2, h = tid & 3;
    float s0 = Sc[row][0][h], s1 = Sc[row][1][h];
    float s2 = Sc[row][2][h], s3 = Sc[row][3][h];
    float mx = fmaxf(fmaxf(s0, s1), fmaxf(s2, s3));
    float e0 = __expf(s0 - mx), e1 = __expf(s1 - mx);
    float e2 = __expf(s2 - mx), e3 = __expf(s3 - mx);
    float inv = 1.f / (e0 + e1 + e2 + e3);
    Wb[row][0][h] = e0 * inv; Wb[row][1][h] = e1 * inv;
    Wb[row][2][h] = e2 * inv; Wb[row][3][h] = e3 * inv;
  }
  __syncthreads();

  // ---- V pass: out += w[t] * (T_t x Wv^T); hoisted Wv frags, V in regs ----
  {
    v16h bv[2][4];
#pragma unroll
    for (int i = 0; i < 2; ++i)
#pragma unroll
      for (int k = 0; k < 4; ++k)
        bv[i][k] = load_bfrag(Wv_h + (size_t)((wave * 2 + i) * 16 + lm) * C_IN +
                              k * 32 + half * 16);

    v8f oacc[MT][2];
    oacc[0][0] = {}; oacc[0][1] = {}; oacc[1][0] = {}; oacc[1][1] = {};
    const int h = lm & 3;               // column o = nt*16+lm -> head = o % 4
    for (int t = 0; t < TPL; ++t) {
      v16h at[MT][4];
#pragma unroll
      for (int mt = 0; mt < MT; ++mt)
#pragma unroll
        for (int k = 0; k < 4; ++k)
          at[mt][k] = load_afrag(Ts[t] + mt * 16 * ZST, ZST, lm, k * 32, half);
#pragma unroll
      for (int i = 0; i < 2; ++i)
#pragma unroll
        for (int mt = 0; mt < MT; ++mt) {
          v8f vac = {};
#pragma unroll
          for (int k = 0; k < 4; ++k) vac = wmma16(at[mt][k], bv[i][k], vac);
#pragma unroll
          for (int v = 0; v < 8; ++v)
            oacc[mt][i][v] += Wb[mt * 16 + v + half * 8][t][h] * vac[v];
        }
    }
#pragma unroll
    for (int mt = 0; mt < MT; ++mt)
#pragma unroll
      for (int i = 0; i < 2; ++i) {
        int nt = wave * 2 + i;
#pragma unroll
        for (int v = 0; v < 8; ++v)
          Os[(mt * 16 + v + half * 8) * OST + nt * 16 + lm] = (_Float16)oacc[mt][i][v];
      }
  }
  __syncthreads();

  // ---- GEMM2: out (32x128) = Out (32x256) x Wo^T + bo; 1 n-tile per wave ----
  {
    const int nt = wave;                 // 8 waves x 16 cols = 128 outputs
    const float bias = bo[nt * 16 + lm]; // C fragment is column-constant
#pragma unroll
    for (int mt = 0; mt < MT; ++mt) {    // sequential m-tiles: lower VGPR load
      v16h ao[8];
#pragma unroll
      for (int k = 0; k < 8; ++k)
        ao[k] = load_afrag(Os + mt * 16 * OST, OST, lm, k * 32, half);
      v8f acc;
#pragma unroll
      for (int v = 0; v < 8; ++v) acc[v] = bias;
#pragma unroll
      for (int k = 0; k < 8; ++k) {
        v16h b = load_bfrag(Wo_h + (size_t)(nt * 16 + lm) * C_QK + k * 32 + half * 16);
        acc = wmma16(ao[k], b, acc);
      }
#pragma unroll
      for (int v = 0; v < 8; ++v)
        out[(size_t)(p0 + mt * 16 + v + half * 8) * C_IN + nt * 16 + lm] = acc[v];
    }
  }
}

// ---------------------------------------------------------------------------
extern "C" void kernel_launch(void* const* d_in, const int* in_sizes, int n_in,
                              void* d_out, int out_size, void* d_ws, size_t ws_size,
                              hipStream_t stream) {
  (void)in_sizes; (void)n_in; (void)out_size; (void)ws_size;
  const float* z2d = (const float*)d_in[0];
  const float* t2d = (const float*)d_in[1];
  const float* Wq  = (const float*)d_in[2];
  const float* Wk  = (const float*)d_in[3];
  const float* Wv  = (const float*)d_in[4];
  const float* Wo  = (const float*)d_in[5];
  const float* bo  = (const float*)d_in[6];
  float* out = (float*)d_out;

  _Float16* wsh = (_Float16*)d_ws;       // 4 x 32768 f16 = 256 KB scratch
  const int NW = C_QK * C_IN;            // 32768 elements per weight matrix
  cvt_f32_f16<<<NW / 256, 256, 0, stream>>>(Wq, wsh,          NW);
  cvt_f32_f16<<<NW / 256, 256, 0, stream>>>(Wk, wsh + NW,     NW);
  cvt_f32_f16<<<NW / 256, 256, 0, stream>>>(Wv, wsh + 2 * NW, NW);
  cvt_f32_f16<<<NW / 256, 256, 0, stream>>>(Wo, wsh + 3 * NW, NW);

  tpa_main<<<NPOS / ROWS, 256, 0, stream>>>(z2d, t2d, wsh, wsh + NW,
                                            wsh + 2 * NW, wsh + 3 * NW, bo, out);
}
